// SelfAttention_28724741275973
// MI455X (gfx1250) — matrix operand order repaired and checked by reading, not compiled, not run on hardware
//
#include <hip/hip_runtime.h>

// ---------------------------------------------------------------------------
// Self-attention, MI455X (gfx1250, wave32).
//  - all matmuls: v_wmma_f32_16x16x32_bf16
//  - K tiles: TDM tensor_load_to_lds (TENSORcnt);  V/mask/X tiles:
//    global_load_async_to_lds_b128 (ASYNCcnt); all double-buffered
//  - native v_cvt_pk_bf16_f32 converts
// ---------------------------------------------------------------------------

typedef __attribute__((ext_vector_type(8)))  float          v8f;
typedef __attribute__((ext_vector_type(16))) __bf16         v16bf;
typedef __attribute__((ext_vector_type(16))) unsigned short v16u;
typedef __attribute__((ext_vector_type(8)))  unsigned short v8u;
typedef __attribute__((ext_vector_type(4)))  unsigned int   u32x4;
typedef __attribute__((ext_vector_type(8)))  int            i32x8;
typedef __attribute__((ext_vector_type(4)))  int            i32x4;

union BFrag { v16u u; v16bf b; };

static constexpr int Bn = 8, S = 2048, E = 1024, H = 64;
static constexpr int WF_ELEMS = 32 * 4 * 32 * 16;  // 65536 halves per projection

#if __has_builtin(__builtin_amdgcn_tensor_load_to_lds)
#define SA_TDM 1
#else
#define SA_TDM 0
#endif

__device__ __forceinline__ v8f zero8() {
  v8f z;
  #pragma unroll
  for (int i = 0; i < 8; ++i) z[i] = 0.0f;
  return z;
}

__device__ __forceinline__ v8f bfwmma(const BFrag& a, const BFrag& b, v8f c) {
  return __builtin_amdgcn_wmma_f32_16x16x32_bf16(
      false, a.b, false, b.b, (short)0, c, false, false);
}

// one 16-byte async copy: global memory -> LDS (tracked by ASYNCcnt)
__device__ __forceinline__ void async_cp16(unsigned lds_off, const void* gp) {
  asm volatile("global_load_async_to_lds_b128 %0, %1, off"
               :: "v"(lds_off), "v"((unsigned long long)(size_t)gp)
               : "memory");
}
__device__ __forceinline__ void wait_async_le5() {
  asm volatile("s_wait_asynccnt 0x5" ::: "memory");
}
__device__ __forceinline__ void wait_async_le4() {
  asm volatile("s_wait_asynccnt 0x4" ::: "memory");
}
__device__ __forceinline__ void wait_async_le3() {
  asm volatile("s_wait_asynccnt 0x3" ::: "memory");
}
__device__ __forceinline__ void wait_async_0() {
  asm volatile("s_wait_asynccnt 0x0" ::: "memory");
}

#if SA_TDM
// TDM: 2-D tile load, 32 rows x 64 bf16 (contiguous 4KB) global -> LDS.
// D# group0: count=1 | lds_addr | global_addr(57b) | type=2
// D# group1: data_size=2B; tensor_dim0=64, dim1=32; tile 64x32; stride0=64
__device__ __forceinline__ void tdm_load_k32x64(unsigned lds_off, const void* gp) {
  unsigned long long ga = (unsigned long long)(size_t)gp;
  u32x4 g0;
  g0[0] = 1u;                                               // count=1
  g0[1] = lds_off;                                          // lds_addr
  g0[2] = (unsigned)(ga & 0xFFFFFFFFull);                   // global_addr lo
  g0[3] = (unsigned)((ga >> 32) & 0x01FFFFFFull) | (2u << 30); // addr hi | type=2
  i32x8 g1;
  g1[0] = 0x10000;      // workgroup_mask=0, data_size=1 (2 bytes)
  g1[1] = (64 << 16);   // tensor_dim0 = 64  (bits 79:48, low half)
  g1[2] = (32 << 16);   // tensor_dim1 = 32  (bits 111:80, low half)
  g1[3] = (64 << 16);   // tile_dim0  = 64   (bits 127:112)
  g1[4] = 32;           // tile_dim1  = 32   (bits 143:128)
  g1[5] = 64;           // tensor_dim0_stride = 64 (bits 207:160, low 32)
  g1[6] = 0;
  g1[7] = 0;
  i32x4 z4 = {0, 0, 0, 0};
#if __clang_major__ >= 23
  i32x8 z8 = {0, 0, 0, 0, 0, 0, 0, 0};
  __builtin_amdgcn_tensor_load_to_lds(g0, g1, z4, z4, z8, 0);
#else
  __builtin_amdgcn_tensor_load_to_lds(g0, g1, z4, z4, 0);
#endif
}
#endif

// ---------------------------------------------------------------------------
// Kernel 1: repack W[E,H] f32 -> per-lane WMMA B-fragment layout (bf16).
// ---------------------------------------------------------------------------
__global__ void __launch_bounds__(256)
sa_prep_w(const float* __restrict__ Wq, const float* __restrict__ Wk,
          const float* __restrict__ Wv, __bf16* __restrict__ wfrag) {
  int tid = blockIdx.x * blockDim.x + threadIdx.x;
  if (tid >= 3 * WF_ELEMS) return;
  int p   = tid >> 16;
  int idx = tid & (WF_ELEMS - 1);
  int t = idx & 15, l = (idx >> 4) & 31, j = (idx >> 9) & 3, s = idx >> 11;
  int k = s * 32 + ((l < 16) ? 0 : 8) + (t & 7) + ((t & 8) << 1);
  int n = j * 16 + (l & 15);
  const float* W = (p == 0) ? Wq : (p == 1) ? Wk : Wv;
  wfrag[tid] = (__bf16)W[k * H + n];
}

// ---------------------------------------------------------------------------
// Kernel 2: projections. Block = 4 waves = 64 rows. Per K-step the 64x32 f32
// X tile is async-copied to LDS (double buffered); W B-fragments are
// double-buffered in VGPRs so loads overlap WMMA.
// Q,K stored bf16 [B*S,H]; V stored transposed bf16 [B,H,S].
// ---------------------------------------------------------------------------
__global__ void __launch_bounds__(128)
sa_proj(const float* __restrict__ Xq, const float* __restrict__ Xk,
        const float* __restrict__ Xv,
        const __bf16* __restrict__ wfrag,
        const float* __restrict__ bq, const float* __restrict__ bk,
        const float* __restrict__ bv,
        unsigned short* __restrict__ Qp, unsigned short* __restrict__ Kp,
        unsigned short* __restrict__ Vt) {
  const int tid  = threadIdx.x;
  const int lane = tid & 31, wave = tid >> 5;
  const int proj = blockIdx.y;
  const int rowbase0 = blockIdx.x * 64;           // first row of the block
  const int rowbase  = rowbase0 + wave * 16;      // this wave's 16-row tile

  const float* X    = (proj == 0) ? Xq : (proj == 1) ? Xk : Xv;
  const float* bias = (proj == 0) ? bq : (proj == 1) ? bk : bv;
  const __bf16* Wf  = wfrag + proj * WF_ELEMS;

  __shared__ __align__(32) float xbuf[2][64 * 32];          // 16 KB ping-pong
  const unsigned xoff = (unsigned)(size_t)&xbuf[0][0];

  // async-copy stage s: 64 rows x 32 f32 (each thread 64B = 4x b128)
  const int crow = tid >> 1, cch = (tid & 1) * 16;          // 16 floats
  auto issue = [&](int s) {
    const char* src = (const char*)(X + (size_t)(rowbase0 + crow) * E + s * 32 + cch);
    unsigned    dst = xoff + ((s & 1) * 64 * 32 + crow * 32 + cch) * 4;
    async_cp16(dst,      src);
    async_cp16(dst + 16, src + 16);
    async_cp16(dst + 32, src + 32);
    async_cp16(dst + 48, src + 48);
  };

  v8f acc[4];
  #pragma unroll
  for (int j = 0; j < 4; ++j) acc[j] = zero8();

  const int hsel = (lane < 16) ? 0 : 8;  // A-frag: lane<16 K=0..7/16..23 else 8..15/24..31
  const int ln   = lane & 15;

  // preload B fragments for step 0 (double buffered across steps)
  BFrag bm[4];
  #pragma unroll
  for (int j = 0; j < 4; ++j)
    bm[j].u = *(const v16u*)(Wf + (size_t)(j * 32 + lane) * 16);

  issue(0);
  for (int s = 0; s < 32; ++s) {
    BFrag bn[4];
    if (s + 1 < 32) {
      #pragma unroll
      for (int j = 0; j < 4; ++j)   // next step's B fragments: overlap WMMA
        bn[j].u = *(const v16u*)(Wf + (size_t)(((s + 1) * 4 + j) * 32 + lane) * 16);
      issue(s + 1);
      wait_async_le4();
    } else {
      wait_async_0();
    }
    __syncthreads();                                   // stage s visible to all

    const float* ap = &xbuf[s & 1][(wave * 16 + ln) * 32 + hsel];
    BFrag a;
    #pragma unroll
    for (int t = 0; t < 8; ++t) {
      a.b[t]     = (__bf16)ap[t];
      a.b[8 + t] = (__bf16)ap[16 + t];
    }
    __syncthreads();                                   // reads done before reuse

    #pragma unroll
    for (int j = 0; j < 4; ++j) acc[j] = bfwmma(a, bm[j], acc[j]);
    if (s + 1 < 32) {
      #pragma unroll
      for (int j = 0; j < 4; ++j) bm[j] = bn[j];
    }
  }

  // C layout: VGPR r -> row (lane<16 ? r : 8+r), col = lane&15
  #pragma unroll
  for (int j = 0; j < 4; ++j) {
    int n = j * 16 + ln;
    float bb = bias[n];
    #pragma unroll
    for (int r = 0; r < 8; ++r) {
      int m = rowbase + ((lane < 16) ? r : 8 + r);
      __bf16 val = (__bf16)(acc[j][r] + bb);
      unsigned short bits = __builtin_bit_cast(unsigned short, val);
      if (proj == 0)      Qp[(size_t)m * H + n] = bits;
      else if (proj == 1) Kp[(size_t)m * H + n] = bits;
      else {
        int bidx = m >> 11, sl = m & (S - 1);
        Vt[((size_t)bidx * H + n) * S + sl] = bits;    // transposed store
      }
    }
  }
}

// ---------------------------------------------------------------------------
// Kernel 3: fused flash attention. Block = 4 waves x 16 query rows. Per
// 32-key block: K tile (4KB) via TDM, V (4KB) + mask (2KB) via async-to-LDS,
// double buffered and shared by all 4 waves. 8 WMMAs per block per wave.
// ---------------------------------------------------------------------------
__global__ void __launch_bounds__(128)
sa_attn(const unsigned short* __restrict__ Qp, const unsigned short* __restrict__ Kp,
        const unsigned short* __restrict__ Vt, const unsigned char* __restrict__ mask,
        float* __restrict__ out) {
  const int tid  = threadIdx.x;
  const int lane = tid & 31, wave = tid >> 5;
  const int b  = blockIdx.y;
  const int q0 = blockIdx.x * 64;                 // block's first query row
  const int qbase = q0 + wave * 16;
  const int half = (lane < 16) ? 0 : 1;
  const int ln = lane & 15;
  const int NKB = S / 32;

  // LDS: K[2][32x64]h | V[2][64x32]h | M[2][64x32]B | P[4][16][32]bf16
  __shared__ __align__(32) unsigned char smem[8192 + 8192 + 4096 + 4096];
  unsigned short* ldsK = (unsigned short*)smem;
  unsigned short* ldsV = (unsigned short*)(smem + 8192);
  unsigned char*  ldsM = smem + 16384;
  __bf16*         plds = (__bf16*)(smem + 20480);
  const unsigned base = (unsigned)(size_t)smem;

  // V + mask async stage: 3 x b128 per thread
  auto issue_vm = [&](int kb) {
    const int buf = kb & 1;
    {   // V tile: 64 h-rows x 64B from Vt (stride S halves)
      int row = tid >> 1, ch = (tid & 1) * 32;
      const char* src = (const char*)(Vt + (size_t)(b * H + row) * S + kb * 32) + ch;
      unsigned    dst = base + 8192 + buf * 4096 + row * 64 + ch;
      async_cp16(dst, src); async_cp16(dst + 16, src + 16);
    }
    {   // mask tile: 64 q-rows x 32B
      int row = tid >> 1, ch = (tid & 1) * 16;
      const unsigned char* src = mask + (size_t)(b * S + q0 + row) * S + kb * 32 + ch;
      unsigned dst = base + 16384 + buf * 2048 + row * 32 + ch;
      async_cp16(dst, src);
    }
  };
  // K tile stage
  auto issue_k = [&](int kb) {
    const int buf = kb & 1;
#if SA_TDM
    if (wave == 0)
      tdm_load_k32x64(base + buf * 4096, Kp + (size_t)(b * S + kb * 32) * H);
#else
    const char* src = (const char*)(Kp + (size_t)(b * S + kb * 32) * H) + tid * 32;
    unsigned    dst = base + buf * 4096 + tid * 32;
    async_cp16(dst, src); async_cp16(dst + 16, src + 16);
#endif
  };

  // --- Q A-fragments for h-steps k0 = 0, 32 (Q stays in VGPRs) ---
  BFrag qa[2];
  {
    const unsigned short* qrow = Qp + (size_t)(b * S + qbase + ln) * H;
    #pragma unroll
    for (int s = 0; s < 2; ++s) {
      const unsigned short* p0 = qrow + s * 32 + half * 16;
      v8u lo = *(const v8u*)(p0);
      v8u hi = *(const v8u*)(p0 + 8);
      #pragma unroll
      for (int t = 0; t < 8; ++t) { qa[s].u[t] = lo[t]; qa[s].u[8 + t] = hi[t]; }
    }
  }

  float mrow[8], lrow[8];
  #pragma unroll
  for (int r = 0; r < 8; ++r) { mrow[r] = -1e30f; lrow[r] = 0.0f; }
  v8f acc[4];
  #pragma unroll
  for (int j = 0; j < 4; ++j) acc[j] = zero8();

  issue_k(0);
  issue_vm(0);
  for (int kb = 0; kb < NKB; ++kb) {
    const int buf = kb & 1;
    if (kb + 1 < NKB) {
      issue_k(kb + 1);
      issue_vm(kb + 1);
#if SA_TDM
      wait_async_le3();
      if (wave == 0) __builtin_amdgcn_s_wait_tensorcnt(1);
#else
      wait_async_le5();
#endif
    } else {
      wait_async_0();
#if SA_TDM
      if (wave == 0) __builtin_amdgcn_s_wait_tensorcnt(0);
#endif
    }
    __syncthreads();                                  // stage kb landed

    // ---- read all tile data for this stage from LDS ----
    BFrag kf[2][2];
    #pragma unroll
    for (int c = 0; c < 2; ++c)
      #pragma unroll
      for (int s = 0; s < 2; ++s)
        kf[c][s].u = *(const v16u*)(ldsK + buf * 2048 +
                                    (c * 16 + ln) * 64 + s * 32 + half * 16);
    BFrag vb[4];
    #pragma unroll
    for (int j = 0; j < 4; ++j)
      vb[j].u = *(const v16u*)(ldsV + buf * 2048 + (j * 16 + ln) * 32 + half * 16);
    unsigned char mk0[8], mk1[8];
    #pragma unroll
    for (int r = 0; r < 8; ++r) {
      const unsigned char* mp =
          ldsM + buf * 2048 + (wave * 16 + half * 8 + r) * 32;
      mk0[r] = mp[ln];
      mk1[r] = mp[16 + ln];
    }

    // ---- scores: Q @ K^T ----
    v8f sc0 = zero8(), sc1 = zero8();
    sc0 = bfwmma(qa[0], kf[0][0], sc0);
    sc0 = bfwmma(qa[1], kf[0][1], sc0);
    sc1 = bfwmma(qa[0], kf[1][0], sc1);
    sc1 = bfwmma(qa[1], kf[1][1], sc1);

    // ---- masked-fill(1e-10) + online softmax over 32 columns ----
    float p0r[8], p1r[8];
    #pragma unroll
    for (int r = 0; r < 8; ++r) {
      float s0 = mk0[r] ? 1e-10f : sc0[r] * 0.125f;   // 1/sqrt(64)
      float s1 = mk1[r] ? 1e-10f : sc1[r] * 0.125f;
      float rmax = fmaxf(s0, s1);
      #pragma unroll
      for (int off = 8; off >= 1; off >>= 1)
        rmax = fmaxf(rmax, __shfl_xor(rmax, off, 16));
      float mnew  = fmaxf(mrow[r], rmax);
      float alpha = __expf(mrow[r] - mnew);
      float p0 = __expf(s0 - mnew), p1 = __expf(s1 - mnew);
      float rsum = p0 + p1;
      #pragma unroll
      for (int off = 8; off >= 1; off >>= 1)
        rsum += __shfl_xor(rsum, off, 16);
      lrow[r] = lrow[r] * alpha + rsum;
      mrow[r] = mnew;
      #pragma unroll
      for (int j = 0; j < 4; ++j) acc[j][r] *= alpha;
      p0r[r] = p0; p1r[r] = p1;
    }

    // ---- transpose P (C layout) -> A-fragment layout via LDS (bf16) ----
    #pragma unroll
    for (int r = 0; r < 8; ++r) {
      int row = half * 8 + r;
      plds[(wave * 16 + row) * 32 + ln]      = (__bf16)p0r[r];
      plds[(wave * 16 + row) * 32 + 16 + ln] = (__bf16)p1r[r];
    }
    __syncthreads();                    // also fences tile reads vs next issue
    BFrag pa;
    {
      const __bf16* pr = plds + (wave * 16 + ln) * 32 + half * 16;
      v8u lo = *(const v8u*)(pr);
      v8u hi = *(const v8u*)(pr + 8);
      #pragma unroll
      for (int t = 0; t < 8; ++t) { pa.u[t] = lo[t]; pa.u[8 + t] = hi[t]; }
    }

    // ---- P @ V ----
    #pragma unroll
    for (int j = 0; j < 4; ++j) acc[j] = bfwmma(pa, vb[j], acc[j]);
  }

  // ---- normalize and store out[B,S,H] f32 ----
  #pragma unroll
  for (int j = 0; j < 4; ++j) {
    int n = j * 16 + ln;
    #pragma unroll
    for (int r = 0; r < 8; ++r) {
      int q = qbase + half * 8 + r;
      out[(size_t)(b * S + q) * H + n] = acc[j][r] / lrow[r];
    }
  }
}

// ---------------------------------------------------------------------------
extern "C" void kernel_launch(void* const* d_in, const int* in_sizes, int n_in,
                              void* d_out, int out_size, void* d_ws, size_t ws_size,
                              hipStream_t stream) {
  const float*         inQ  = (const float*)d_in[0];
  const float*         inK  = (const float*)d_in[1];
  const float*         inV  = (const float*)d_in[2];
  const unsigned char* mask = (const unsigned char*)d_in[3];  // bool, 1 byte
  const float* Wq = (const float*)d_in[4];
  const float* bq = (const float*)d_in[5];
  const float* Wk = (const float*)d_in[6];
  const float* bk = (const float*)d_in[7];
  const float* Wv = (const float*)d_in[8];
  const float* bv = (const float*)d_in[9];

  // Workspace: Wfrags 384K | Qp 2M | Kp 2M | Vt 2M  (~6.7 MB, L2-resident)
  unsigned char* ws = (unsigned char*)d_ws;
  __bf16*         wfrag = (__bf16*)(ws);
  unsigned short* Qp    = (unsigned short*)(ws + 393216);
  unsigned short* Kp    = (unsigned short*)(ws + 393216 + 2097152);
  unsigned short* Vt    = (unsigned short*)(ws + 393216 + 2 * 2097152);

  sa_prep_w<<<dim3(768), dim3(256), 0, stream>>>(Wq, Wk, Wv, wfrag);
  sa_proj<<<dim3((Bn * S / 16) / 4, 3), dim3(128), 0, stream>>>(
      inQ, inK, inV, wfrag, bq, bk, bv, Qp, Kp, Vt);
  sa_attn<<<dim3(S / 64, Bn), dim3(128), 0, stream>>>(
      Qp, Kp, Vt, mask, (float*)d_out);
}
